// APPNP_4303557231208
// MI455X (gfx1250) — matrix operand order repaired and checked by reading, not compiled, not run on hardware
//
#include <hip/hip_runtime.h>

typedef __attribute__((ext_vector_type(16))) __bf16 v16bf;
typedef __attribute__((ext_vector_type(8)))  float  v8f;

#define N_NODES 100000
#define F_IN    512
#define H_HID   256
#define C_OUT   64
#define ALPHA   0.1f

// ---------------- TDM support (guarded; falls back to manual staging) ----------------
#if defined(__gfx1250__) && __has_builtin(__builtin_amdgcn_tensor_load_to_lds) && \
    __has_builtin(__builtin_amdgcn_s_wait_tensorcnt)
#define USE_TDM 1
typedef unsigned int tdm_v4u __attribute__((ext_vector_type(4)));
typedef int          tdm_v8i __attribute__((ext_vector_type(8)));
typedef int          tdm_v4i __attribute__((ext_vector_type(4)));

// low 32 bits of a generic pointer to LDS == LDS byte offset (aperture model)
__device__ __forceinline__ unsigned lds_off(const void* p) {
    return (unsigned)(unsigned long long)p;
}

// 2D bf16 tile load: tensor [d1 rows x d0 elems], row stride d0_stride elems,
// tile [tile_d1 x tile_d0], LDS rows padded via pad_interval/pad_amount codes.
__device__ __forceinline__ void tdm_load_2d_bf16(
    unsigned lds_byte_off, const void* gaddr,
    unsigned tensor_d0, unsigned tensor_d1,
    unsigned tile_d0, unsigned tile_d1,
    unsigned d0_stride_elems,
    unsigned pad_interval_code, unsigned pad_amount_code) {
    unsigned long long ga = (unsigned long long)gaddr;
    tdm_v4u g0;
    g0[0] = 1u;                                                  // count = 1 (valid D#)
    g0[1] = lds_byte_off;                                        // lds_addr
    g0[2] = (unsigned)ga;                                        // global_addr[31:0]
    g0[3] = (unsigned)((ga >> 32) & 0x01FFFFFFu) | (2u << 30);   // global_addr[56:32] | type=2
    tdm_v8i g1;
    g1[0] = (int)((1u << 16) |                                   // data_size = 2 bytes
                  (1u << 20) |                                   // pad_enable
                  (pad_interval_code << 22) |
                  (pad_amount_code << 25));
    g1[1] = (int)((tensor_d0 & 0xFFFFu) << 16);                  // tensor_dim0[15:0]
    g1[2] = (int)((tensor_d0 >> 16) | ((tensor_d1 & 0xFFFFu) << 16));
    g1[3] = (int)((tensor_d1 >> 16) | (tile_d0 << 16));          // tile_dim0
    g1[4] = (int)(tile_d1 & 0xFFFFu);                            // tile_dim1 (tile_dim2 = 0)
    g1[5] = (int)d0_stride_elems;                                // tensor_dim0_stride[31:0]
    g1[6] = 0;
    g1[7] = 0;
    tdm_v4i gz = {};
#if __has_include(<hip/amd_detail/amd_gfx1250_TDM.h>)
    tdm_v8i gz8 = {};
    __builtin_amdgcn_tensor_load_to_lds(g0, g1, gz, gz, gz8, 0); // clang-23 arity
#else
    __builtin_amdgcn_tensor_load_to_lds(g0, g1, gz, gz, 0);      // ROCm 7.2 arity
#endif
}
#else
#define USE_TDM 0
#endif

// ---------------- utility kernels ----------------
__global__ void fill_f32(float* __restrict__ p, long n, float v) {
    long i = (long)blockIdx.x * blockDim.x + threadIdx.x;
    if (i < n) p[i] = v;
}

// W1 [F,H] f32 row-major -> W1T [H,F] bf16
__global__ void conv_w1(const float* __restrict__ W1, __bf16* __restrict__ W1T) {
    int i = blockIdx.x * blockDim.x + threadIdx.x;
    if (i >= F_IN * H_HID) return;
    int k = i / H_HID, n = i % H_HID;
    W1T[n * F_IN + k] = (__bf16)W1[i];
}

// W2 [H,C] f32 -> W2T [C,H] bf16 with variational-dropout retain prob folded in
__global__ void conv_w2(const float* __restrict__ W2, const float* __restrict__ p,
                        __bf16* __restrict__ W2T) {
    int i = blockIdx.x * blockDim.x + threadIdx.x;
    if (i >= H_HID * C_OUT) return;
    int k = i / C_OUT, n = i % C_OUT;
    float pr = fminf(fmaxf(p[k], 0.f), 1.f);
    W2T[n * H_HID + k] = (__bf16)(W2[i] * pr);
}

// ---------------- GEMM1: h = relu(x @ W1 + b1), out bf16 [N,H] ----------------
// WG = 256 threads = 8 waves; each wave: 16 rows x 256 cols (16 wmma accumulators).
// W1T k-slabs double-buffered in LDS via TDM (row stride padded 40 elems = 80B).
__global__ __launch_bounds__(256) void gemm1_kernel(
    const float* __restrict__ x, const __bf16* __restrict__ W1T,
    const float* __restrict__ b1, __bf16* __restrict__ hh) {
    __shared__ __bf16 sB[2][H_HID * 40];
    const int tid  = threadIdx.x;
    const int lane = tid & 31;
    const int wave = tid >> 5;
    const int rowbase = (blockIdx.x * 8 + wave) * 16;
    int ar = rowbase + (lane & 15);
    if (ar > N_NODES - 1) ar = N_NODES - 1;          // clamp loads, keep EXEC all-1 for WMMA
    const int kha = (lane < 16) ? 0 : 8;             // A-fragment K base (16-bit A layout)
    const int khb = (lane < 16) ? 0 : 16;            // unused; B fragment uses kha (same K order as A)

    v8f acc[16];
#pragma unroll
    for (int j = 0; j < 16; ++j) acc[j] = {};

#if USE_TDM
    if (tid < 32)   // one wave issues; 64B rows -> interval code 3, pad 4 DWORDs -> code 3
        tdm_load_2d_bf16(lds_off(&sB[0][0]), W1T, F_IN, H_HID, 32, H_HID, F_IN, 3, 3);
#endif

    const int KSTEPS = F_IN / 32;
    for (int step = 0; step < KSTEPS; ++step) {
        const int kb = step * 32;
        const __bf16* sbuf;
#if USE_TDM
        sbuf = &sB[step & 1][0];
        __builtin_amdgcn_s_wait_tensorcnt(0);        // no-op for non-issuing waves
        __syncthreads();                             // slab ready; prev compute done
        if (tid < 32 && step + 1 < KSTEPS)           // prefetch next slab into other buffer
            tdm_load_2d_bf16(lds_off(&sB[(step + 1) & 1][0]), W1T + (kb + 32),
                             F_IN, H_HID, 32, H_HID, F_IN, 3, 3);
#else
        sbuf = &sB[0][0];
        {   // manual stage: 256 threads, one W1T row each
            const uint4* g = (const uint4*)(W1T + tid * F_IN + kb);
            uint4* s = (uint4*)(&sB[0][0] + tid * 40);
#pragma unroll
            for (int c = 0; c < 4; ++c) s[c] = g[c];
        }
        __syncthreads();
#endif
        // A fragment: rows striped per lane, K = kha..kha+7 and kha+16..kha+23
        const float* xr = x + (long)ar * F_IN + kb + kha;
        float af[16];
        *(float4*)(af + 0)  = *(const float4*)(xr + 0);
        *(float4*)(af + 4)  = *(const float4*)(xr + 4);
        *(float4*)(af + 8)  = *(const float4*)(xr + 16);
        *(float4*)(af + 12) = *(const float4*)(xr + 20);
        if (kb + 32 < F_IN) __builtin_prefetch(xr + 32, 0, 0);   // global_prefetch_b8
        v16bf a;
#pragma unroll
        for (int i = 0; i < 16; ++i) a[i] = (__bf16)af[i];

#pragma unroll
        for (int j = 0; j < 16; ++j) {
            const uint4* bp = (const uint4*)(sbuf + (j * 16 + (lane & 15)) * 40 + kha);
            v16bf b;
            uint4* pb = (uint4*)&b;
            pb[0] = bp[0];
            pb[1] = bp[2];
            acc[j] = __builtin_amdgcn_wmma_f32_16x16x32_bf16(
                false, a, false, b, (short)0, acc[j], false, false);
        }
#if !USE_TDM
        __syncthreads();
#endif
    }

    // epilogue: +bias, relu, bf16 store (C layout: VGPR v -> row v / v+8 per half-wave)
#pragma unroll
    for (int j = 0; j < 16; ++j) {
        int col = j * 16 + (lane & 15);
        float bias = b1[col];
#pragma unroll
        for (int v = 0; v < 8; ++v) {
            int row = rowbase + ((lane < 16) ? v : v + 8);
            if (row < N_NODES) {
                float val = acc[j][v] + bias;
                hh[(long)row * H_HID + col] = (__bf16)fmaxf(val, 0.f);
            }
        }
    }
}

// ---------------- GEMM2: h0 = hh @ W2p + b2, out f32 [N,C] ----------------
__global__ __launch_bounds__(256) void gemm2_kernel(
    const __bf16* __restrict__ hh, const __bf16* __restrict__ W2T,
    const float* __restrict__ b2, float* __restrict__ h0) {
    __shared__ __bf16 sB[C_OUT * 264];   // padded stride 264 elems (528B rows)
    const int tid  = threadIdx.x;
    const int lane = tid & 31;
    const int wave = tid >> 5;
    const int rowbase = (blockIdx.x * 8 + wave) * 16;
    int ar = rowbase + (lane & 15);
    if (ar > N_NODES - 1) ar = N_NODES - 1;
    const int kha = (lane < 16) ? 0 : 8;
    const int khb = (lane < 16) ? 0 : 16;

#if USE_TDM
    if (tid < 32)   // whole 32KB W2T in one TDM; 512B rows -> interval code 6, pad code 3
        tdm_load_2d_bf16(lds_off(&sB[0]), W2T, H_HID, C_OUT, H_HID, C_OUT, H_HID, 6, 3);
    __builtin_amdgcn_s_wait_tensorcnt(0);
    __syncthreads();
#else
    {   // manual stage: 256 threads, 128B each
        int n = tid >> 2, q = tid & 3;
        const uint4* g = (const uint4*)(W2T + n * H_HID + q * 64);
        uint4* s = (uint4*)(&sB[0] + n * 264 + q * 64);
#pragma unroll
        for (int c = 0; c < 8; ++c) s[c] = g[c];
    }
    __syncthreads();
#endif

    v8f acc[4];
#pragma unroll
    for (int j = 0; j < 4; ++j) acc[j] = {};

    const __bf16* arow = hh + (long)ar * H_HID;
    for (int kb = 0; kb < H_HID; kb += 32) {
        v16bf a;
        uint4* pa = (uint4*)&a;
        pa[0] = *(const uint4*)(arow + kb + kha);
        pa[1] = *(const uint4*)(arow + kb + kha + 16);
#pragma unroll
        for (int j = 0; j < 4; ++j) {
            const uint4* bp = (const uint4*)(&sB[0] + (j * 16 + (lane & 15)) * 264 + kb + kha);
            v16bf b;
            uint4* pb = (uint4*)&b;
            pb[0] = bp[0];
            pb[1] = bp[2];
            acc[j] = __builtin_amdgcn_wmma_f32_16x16x32_bf16(
                false, a, false, b, (short)0, acc[j], false, false);
        }
    }

#pragma unroll
    for (int j = 0; j < 4; ++j) {
        int col = j * 16 + (lane & 15);
        float bias = b2[col];
#pragma unroll
        for (int v = 0; v < 8; ++v) {
            int row = rowbase + ((lane < 16) ? v : v + 8);
            if (row < N_NODES) h0[(long)row * C_OUT + col] = acc[j][v] + bias;
        }
    }
}

// ---------------- graph kernels ----------------
__global__ void degree_k(const int* __restrict__ dst, int E, float* __restrict__ deg) {
    int e = blockIdx.x * blockDim.x + threadIdx.x;
    if (e < E) atomicAdd(deg + dst[e], 1.0f);
}

__global__ void dinv_k(float* __restrict__ deg) {
    int i = blockIdx.x * blockDim.x + threadIdx.x;
    if (i < N_NODES) {
        float d = deg[i];
        deg[i] = (d > 0.f) ? rsqrtf(d) : 0.f;
    }
}

// 16 lanes per edge, float4 per lane: agg[dst] += dinv[src]*dinv[dst]*h[src]
__global__ void gather_k(const int* __restrict__ src, const int* __restrict__ dst, int E,
                         const float* __restrict__ dinv, const float* __restrict__ h,
                         float* __restrict__ agg) {
    long idx = (long)blockIdx.x * blockDim.x + threadIdx.x;
    long e = idx >> 4;
    if (e >= E) return;
    int c = ((int)idx & 15) * 4;
    int s = src[e], d = dst[e];
    float w = dinv[s] * dinv[d];
    float4 v = *(const float4*)(h + (long)s * C_OUT + c);
    float* ap = agg + (long)d * C_OUT + c;
    atomicAdd(ap + 0, w * v.x);
    atomicAdd(ap + 1, w * v.y);
    atomicAdd(ap + 2, w * v.z);
    atomicAdd(ap + 3, w * v.w);
}

// h_next = 0.9*(agg + dinv^2 * h_prev) + 0.1*h0   (self-loop folded in)
__global__ void combine_k(const float* __restrict__ agg, const float* __restrict__ hprev,
                          const float* __restrict__ h0, const float* __restrict__ dinv,
                          float* __restrict__ hnext) {
    long idx = (long)blockIdx.x * blockDim.x + threadIdx.x;
    if (idx >= (long)N_NODES * C_OUT) return;
    int i = (int)(idx >> 6);
    float di = dinv[i];
    hnext[idx] = (1.f - ALPHA) * (agg[idx] + di * di * hprev[idx]) + ALPHA * h0[idx];
}

// ---------------- launch ----------------
extern "C" void kernel_launch(void* const* d_in, const int* in_sizes, int n_in,
                              void* d_out, int out_size, void* d_ws, size_t ws_size,
                              hipStream_t stream) {
    const float* x  = (const float*)d_in[0];
    const float* W1 = (const float*)d_in[1];
    const float* b1 = (const float*)d_in[2];
    const float* p  = (const float*)d_in[3];
    const float* W2 = (const float*)d_in[4];
    const float* b2 = (const float*)d_in[5];
    const int*   ei = (const int*)d_in[6];
    const int E = in_sizes[6] / 2;
    const int* src = ei;
    const int* dst = ei + E;
    float* out = (float*)d_out;

    char* ws = (char*)d_ws;
    size_t off = 0;
    auto carve = [&](size_t bytes) -> char* {
        char* q = ws + off;
        off += (bytes + 255) & ~(size_t)255;
        return q;
    };
    __bf16* W1T = (__bf16*)carve((size_t)F_IN * H_HID * sizeof(__bf16));
    __bf16* W2T = (__bf16*)carve((size_t)H_HID * C_OUT * sizeof(__bf16));
    __bf16* hh  = (__bf16*)carve((size_t)N_NODES * H_HID * sizeof(__bf16));
    float*  h0  = (float*)carve((size_t)N_NODES * C_OUT * sizeof(float));
    float*  hB  = (float*)carve((size_t)N_NODES * C_OUT * sizeof(float));
    float*  agg = (float*)carve((size_t)N_NODES * C_OUT * sizeof(float));
    float*  deg = (float*)carve((size_t)N_NODES * sizeof(float));

    const long NC = (long)N_NODES * C_OUT;
    const int gemmGrid = (N_NODES + 127) / 128;

    conv_w1<<<(F_IN * H_HID + 255) / 256, 256, 0, stream>>>(W1, W1T);
    conv_w2<<<(H_HID * C_OUT + 255) / 256, 256, 0, stream>>>(W2, p, W2T);
    fill_f32<<<(N_NODES + 255) / 256, 256, 0, stream>>>(deg, N_NODES, 1.0f);  // self-loop
    degree_k<<<(E + 255) / 256, 256, 0, stream>>>(dst, E, deg);

    gemm1_kernel<<<gemmGrid, 256, 0, stream>>>(x, W1T, b1, hh);
    gemm2_kernel<<<gemmGrid, 256, 0, stream>>>(hh, W2T, b2, h0);

    dinv_k<<<(N_NODES + 255) / 256, 256, 0, stream>>>(deg);  // deg -> dinv in place

    // APPNP step 1: h_prev = h0
    fill_f32<<<(int)((NC + 255) / 256), 256, 0, stream>>>(agg, NC, 0.f);
    gather_k<<<(int)(((long)E * 16 + 255) / 256), 256, 0, stream>>>(src, dst, E, deg, h0, agg);
    combine_k<<<(int)((NC + 255) / 256), 256, 0, stream>>>(agg, h0, h0, deg, hB);

    // APPNP step 2: h_prev = hB, final write to d_out
    fill_f32<<<(int)((NC + 255) / 256), 256, 0, stream>>>(agg, NC, 0.f);
    gather_k<<<(int)(((long)E * 16 + 255) / 256), 256, 0, stream>>>(src, dst, E, deg, hB, agg);
    combine_k<<<(int)((NC + 255) / 256), 256, 0, stream>>>(agg, hB, h0, deg, out);
}
